// RNNModel_28355374088575
// MI455X (gfx1250) — compile-verified, matrix operand order checked by reading, not run on hardware
//
#include <hip/hip_runtime.h>
#include <hip/hip_bf16.h>
#include <math.h>

// ---------------------------------------------------------------------------
// Problem constants (from reference)
// ---------------------------------------------------------------------------
#define NHID   1024
#define SEQLEN 128
#define BSZ    64
#define NS     10
#define NROWS  (SEQLEN * BSZ)       // 8192
#define TEMP   65.0f
#define CLIPD  0.01f
#define EPS    1e-6f

typedef __bf16 bf16_t;
typedef __attribute__((ext_vector_type(16))) __bf16 v16bf;
typedef __attribute__((ext_vector_type(8)))  __bf16 v8bf;
typedef __attribute__((ext_vector_type(8)))  float  v8f;

// ---------------------------------------------------------------------------
// WMMA helpers (gfx1250 wave32, V_WMMA_F32_16X16X32_BF16)
// ---------------------------------------------------------------------------
__device__ __forceinline__ v8f wmma_bf16(v16bf a, v16bf b, v8f c) {
  return __builtin_amdgcn_wmma_f32_16x16x32_bf16(
      /*neg_a=*/false, a, /*neg_b=*/false, b,
      /*c_mod=*/(short)0, c, /*reuse_a=*/false, /*reuse_b=*/false);
}

// Fragment load from a per-lane base pointer (already offset for this lane's
// row and half-K); two b128 loads at +0 and +32 bytes.  Serves both A (16x32,
// lane row = M) and B (32x16, lane row = N, since B[k][n] = W[n][k]).
// Callers pass "base + compile-time-offset" so K/N advances become immediate
// offsets in the load encoding (no per-iteration 64-bit address math).
__device__ __forceinline__ v16bf ldfrag(const bf16_t* p) {
  v8bf lo = *(const v8bf*)p;          // K = +0 .. +7
  v8bf hi = *(const v8bf*)(p + 16);   // K = +16 .. +23
  v16bf f;
#pragma unroll
  for (int i = 0; i < 8; ++i) { f[i] = lo[i]; f[i + 8] = hi[i]; }
  return f;
}

// ---------------------------------------------------------------------------
// Prep kernels
// ---------------------------------------------------------------------------
__global__ void conv_w_kernel(const float* __restrict__ Wih,
                              const float* __restrict__ Whh,
                              const float* __restrict__ bih,
                              const float* __restrict__ bhh,
                              bf16_t* __restrict__ WihB,
                              bf16_t* __restrict__ WhhB,
                              float* __restrict__ bsum) {
  const int i = blockIdx.x * blockDim.x + threadIdx.x;
  if (i < NHID * NHID) {
    WihB[i] = (bf16_t)Wih[i];
    WhhB[i] = (bf16_t)Whh[i];
  }
  if (i < NHID) bsum[i] = bih[i] + bhh[i];
}

__global__ void embed_kernel(const int* __restrict__ data,
                             const float* __restrict__ embW,
                             bf16_t* __restrict__ Embbf) {
  const int row = blockIdx.x;                    // s*BSZ + b (matches flat data)
  const int tok = data[row];
  const float* src = embW + (size_t)tok * NHID;
  bf16_t* dst = Embbf + (size_t)row * NHID;
  for (int c = threadIdx.x * 4; c < NHID; c += blockDim.x * 4) {
    const float4 v = *(const float4*)(src + c);
    dst[c + 0] = (bf16_t)v.x; dst[c + 1] = (bf16_t)v.y;
    dst[c + 2] = (bf16_t)v.z; dst[c + 3] = (bf16_t)v.w;
  }
}

__global__ void init_h0_kernel(float* __restrict__ rawF,
                               bf16_t* __restrict__ Hbf) {
  const int i = blockIdx.x * blockDim.x + threadIdx.x;
  if (i < BSZ * NHID) { rawF[i] = 0.0f; Hbf[i] = (bf16_t)0.0f; }
}

// ---------------------------------------------------------------------------
// RNN step: H_t = tanh(E_t @ Wih^T + H_{t-1} @ Whh^T + (b_ih+b_hh))
// grid = 32 blocks (N chunks of 32), block = 128 thr = 4 waves (M chunks of 16)
// Ping-pong double buffering (unroll-2, no buffer-rotation copies).
// Wave tile 16x32 keeps total VGPRs ~140 (no spills).
// ---------------------------------------------------------------------------
__global__ void rnn_step_kernel(const bf16_t* __restrict__ Embbf,
                                const bf16_t* __restrict__ Wih,
                                const bf16_t* __restrict__ Whh,
                                const float* __restrict__ bsum,
                                bf16_t* __restrict__ Hbf,
                                float* __restrict__ rawF,
                                int ts) {
  const int w  = threadIdx.x >> 5;
  const int m0 = w * 16;
  const int n0 = blockIdx.x * 32;
  const int tl = threadIdx.x & 31;
  const int lr = tl & 15;
  const int lc = (tl >> 4) << 3;

  // one base pointer per matrix; all frag addresses = base + immediate
  const bf16_t* pae = Embbf + (size_t)ts * BSZ * NHID + (size_t)(m0 + lr) * NHID + lc;
  const bf16_t* pah = Hbf   + (size_t)ts * BSZ * NHID + (size_t)(m0 + lr) * NHID + lc;
  const bf16_t* pbi = Wih + (size_t)(n0 + lr) * NHID + lc;
  const bf16_t* pbh = Whh + (size_t)(n0 + lr) * NHID + lc;

  v8f acc[2] = {};
  v16bf ae0, ah0, bi0[2], bh0[2];
  v16bf ae1, ah1, bi1[2], bh1[2];

#define RNN_LOAD(AE, AH, BI, BH, K)                                        \
  do {                                                                     \
    AE = ldfrag(pae + (K));                                                \
    AH = ldfrag(pah + (K));                                                \
    BI[0] = ldfrag(pbi + (K));                                             \
    BI[1] = ldfrag(pbi + 16 * NHID + (K));                                 \
    BH[0] = ldfrag(pbh + (K));                                             \
    BH[1] = ldfrag(pbh + 16 * NHID + (K));                                 \
  } while (0)

#define RNN_COMP(AE, AH, BI, BH)                                           \
  do {                                                                     \
    acc[0] = wmma_bf16(AE, BI[0], acc[0]);                                 \
    acc[1] = wmma_bf16(AE, BI[1], acc[1]);                                 \
    acc[0] = wmma_bf16(AH, BH[0], acc[0]);                                 \
    acc[1] = wmma_bf16(AH, BH[1], acc[1]);                                 \
  } while (0)

  RNN_LOAD(ae0, ah0, bi0, bh0, 0);
  for (int k0 = 0; k0 < NHID - 64; k0 += 64) {
    RNN_LOAD(ae1, ah1, bi1, bh1, k0 + 32);
    RNN_COMP(ae0, ah0, bi0, bh0);
    RNN_LOAD(ae0, ah0, bi0, bh0, k0 + 64);
    RNN_COMP(ae1, ah1, bi1, bh1);
  }
  RNN_LOAD(ae1, ah1, bi1, bh1, NHID - 32);
  RNN_COMP(ae0, ah0, bi0, bh0);
  RNN_COMP(ae1, ah1, bi1, bh1);
#undef RNN_LOAD
#undef RNN_COMP

  // epilogue: tanh + bias, write f32 (raw) and bf16 (next step A / prev)
  const int mh = (tl >> 4) * 8;
  bf16_t* outB = Hbf  + (size_t)(ts + 1) * BSZ * NHID;
  float*  outF = rawF + (size_t)(ts + 1) * BSZ * NHID;
#pragma unroll
  for (int j = 0; j < 2; ++j) {
    const int n = n0 + j * 16 + lr;
#pragma unroll
    for (int v = 0; v < 8; ++v) {
      const int m = m0 + mh + v;
      const float h = tanhf(acc[j][v] + bsum[n]);
      outF[(size_t)m * NHID + n] = h;
      outB[(size_t)m * NHID + n] = (bf16_t)h;
    }
  }
}

// ---------------------------------------------------------------------------
// hiddens_U = prev @ Whh^T + b_hh   (prev = Hbf rows 0..8191)
// grid = 128*16 blocks, block = 128 thr; wave tile 16x64, ping-pong buffers
// ---------------------------------------------------------------------------
__global__ void hidu_kernel(const bf16_t* __restrict__ Hbf,
                            const bf16_t* __restrict__ Whh,
                            const float* __restrict__ bhh,
                            float* __restrict__ hidU) {
  const int mt = blockIdx.x >> 4;
  const int nt = blockIdx.x & 15;
  const int w  = threadIdx.x >> 5;
  const int m0 = mt * 64 + w * 16;
  const int n0 = nt * 64;
  const int tl = threadIdx.x & 31;
  const int lr = tl & 15;
  const int lc = (tl >> 4) << 3;

  const bf16_t* pa = Hbf + (size_t)(m0 + lr) * NHID + lc;
  const bf16_t* pb = Whh + (size_t)(n0 + lr) * NHID + lc;

  v8f acc[4] = {};
  v16bf a0, b0[4], a1, b1[4];

#define GU_LOAD(A, B, K)                                                   \
  do {                                                                     \
    A = ldfrag(pa + (K));                                                  \
    B[0] = ldfrag(pb + (K));                                               \
    B[1] = ldfrag(pb + 16 * NHID + (K));                                   \
    B[2] = ldfrag(pb + 32 * NHID + (K));                                   \
    B[3] = ldfrag(pb + 48 * NHID + (K));                                   \
  } while (0)

#define GU_COMP(A, B)                                                      \
  do {                                                                     \
    acc[0] = wmma_bf16(A, B[0], acc[0]);                                   \
    acc[1] = wmma_bf16(A, B[1], acc[1]);                                   \
    acc[2] = wmma_bf16(A, B[2], acc[2]);                                   \
    acc[3] = wmma_bf16(A, B[3], acc[3]);                                   \
  } while (0)

  GU_LOAD(a0, b0, 0);
  for (int k0 = 0; k0 < NHID - 64; k0 += 64) {
    GU_LOAD(a1, b1, k0 + 32);
    GU_COMP(a0, b0);
    GU_LOAD(a0, b0, k0 + 64);
    GU_COMP(a1, b1);
  }
  GU_LOAD(a1, b1, NHID - 32);
  GU_COMP(a0, b0);
  GU_COMP(a1, b1);
#undef GU_LOAD
#undef GU_COMP

  const int mh = (tl >> 4) * 8;
#pragma unroll
  for (int j = 0; j < 4; ++j) {
    const int n = n0 + j * 16 + lr;
#pragma unroll
    for (int v = 0; v < 8; ++v)
      hidU[(size_t)(m0 + mh + v) * NHID + n] = acc[j][v] + bhh[n];
  }
}

// ---------------------------------------------------------------------------
// Negative term (dominant GEMM, gather fused):
//   out = tanh(emb_W[samples] @ Wih^T + hiddens_U)
//   dist = clip(||prev - out + eps||^2, 0, 0.01);  negbuf = exp(-dist)
// grid = NS*512 blocks (one per sample x 16-row slab), block = 128 thr
// A staged once in LDS (bf16); B ping-pong double buffered from global.
// ---------------------------------------------------------------------------
__global__ void neg_kernel(const int* __restrict__ samples,
                           const float* __restrict__ embW,
                           const bf16_t* __restrict__ Wih,
                           const float* __restrict__ bih,
                           const float* __restrict__ hidU,
                           const float* __restrict__ prevF,
                           float* __restrict__ negbuf) {
  const int ns   = blockIdx.x >> 9;
  const int mb   = blockIdx.x & 511;
  const int row0 = mb * 16;
  const int* idx = samples + ns * NROWS + row0;

  __shared__ bf16_t As[16 * NHID];   // 32 KB: gathered 16x1024 A slab (bf16)
  __shared__ float  red[4][16];

  for (int e = threadIdx.x * 4; e < 16 * NHID; e += blockDim.x * 4) {
    const int r = e >> 10;
    const int c = e & 1023;
    const float4 v = *(const float4*)(embW + (size_t)idx[r] * NHID + c);
    As[r * NHID + c + 0] = (bf16_t)v.x;
    As[r * NHID + c + 1] = (bf16_t)v.y;
    As[r * NHID + c + 2] = (bf16_t)v.z;
    As[r * NHID + c + 3] = (bf16_t)v.w;
  }
  __syncthreads();

  const int w  = threadIdx.x >> 5;
  const int tl = threadIdx.x & 31;
  const int lr = tl & 15;
  const int lc = (tl >> 4) << 3;
  const int mh = (tl >> 4) * 8;
  const bf16_t* pa = As + lr * NHID + lc;     // LDS-sourced A fragments

  float dsum[8] = {};
  for (int c = 0; c < 4; ++c) {               // each wave covers 256 N columns
    const int n0 = w * 256 + c * 64;
    const bf16_t* pb = Wih + (size_t)(n0 + lr) * NHID + lc;

    v8f acc[4] = {};
    v16bf b0[4], b1[4];

#define NB_LOAD(B, K)                                                      \
  do {                                                                     \
    B[0] = ldfrag(pb + (K));                                               \
    B[1] = ldfrag(pb + 16 * NHID + (K));                                   \
    B[2] = ldfrag(pb + 32 * NHID + (K));                                   \
    B[3] = ldfrag(pb + 48 * NHID + (K));                                   \
  } while (0)

#define NB_COMP(B, K)                                                      \
  do {                                                                     \
    const v16bf a = ldfrag(pa + (K));    /* ds_load, short latency */      \
    acc[0] = wmma_bf16(a, B[0], acc[0]);                                   \
    acc[1] = wmma_bf16(a, B[1], acc[1]);                                   \
    acc[2] = wmma_bf16(a, B[2], acc[2]);                                   \
    acc[3] = wmma_bf16(a, B[3], acc[3]);                                   \
  } while (0)

    NB_LOAD(b0, 0);
    for (int k0 = 0; k0 < NHID - 64; k0 += 64) {
      NB_LOAD(b1, k0 + 32);
      NB_COMP(b0, k0);
      NB_LOAD(b0, k0 + 64);
      NB_COMP(b1, k0 + 32);
    }
    NB_LOAD(b1, NHID - 32);
    NB_COMP(b0, NHID - 64);
    NB_COMP(b1, NHID - 32);
#undef NB_LOAD
#undef NB_COMP

#pragma unroll
    for (int j = 0; j < 4; ++j) {
      const int n = n0 + j * 16 + lr;
#pragma unroll
      for (int v = 0; v < 8; ++v) {
        const int row = row0 + mh + v;
        const float o = tanhf(acc[j][v] + hidU[(size_t)row * NHID + n] + bih[n]);
        const float d = prevF[(size_t)row * NHID + n] - o + EPS;
        dsum[v] += d * d;
      }
    }
  }
  // reduce across the 16 lanes of each half-wave (lane = n, vgpr = m)
#pragma unroll
  for (int v = 0; v < 8; ++v) {
#pragma unroll
    for (int mask = 1; mask < 16; mask <<= 1)
      dsum[v] += __shfl_xor(dsum[v], mask, 32);
  }
  if (lr == 0) {
#pragma unroll
    for (int v = 0; v < 8; ++v) red[w][mh + v] = dsum[v];
  }
  __syncthreads();
  if (threadIdx.x < 16) {
    const int m = threadIdx.x;
    float dist = red[0][m] + red[1][m] + red[2][m] + red[3][m];
    dist = fminf(fmaxf(dist, 0.0f), CLIPD);
    negbuf[ns * NROWS + row0 + m] = expf(-dist);
  }
}

// ---------------------------------------------------------------------------
// Positive term partials: posPart[t] = ||raw[t] - raw[t+1] + eps||^2 (summed)
// ---------------------------------------------------------------------------
__global__ void pos_kernel(const float* __restrict__ rawF,
                           float* __restrict__ posPart) {
  const int t = blockIdx.x;
  const float* a = rawF + (size_t)t * BSZ * NHID;
  const float* b = a + BSZ * NHID;
  float s = 0.0f;
  for (int i = threadIdx.x; i < BSZ * NHID; i += blockDim.x) {
    const float d = a[i] - b[i] + EPS;
    s += d * d;
  }
  __shared__ float sm[256];
  sm[threadIdx.x] = s;
  __syncthreads();
  for (int o = blockDim.x / 2; o > 0; o >>= 1) {
    if (threadIdx.x < o) sm[threadIdx.x] += sm[threadIdx.x + o];
    __syncthreads();
  }
  if (threadIdx.x == 0) posPart[t] = sm[0];
}

// ---------------------------------------------------------------------------
// Final: loss = TEMP*sum(pos)/SEQLEN + sum_r log(sum_ns exp(-dist)/NROWS + eps)
// ---------------------------------------------------------------------------
__global__ void final_kernel(const float* __restrict__ posPart,
                             const float* __restrict__ negbuf,
                             float* __restrict__ out) {
  float s = 0.0f;
  for (int r = threadIdx.x; r < NROWS; r += blockDim.x) {
    float se = 0.0f;
#pragma unroll
    for (int k = 0; k < NS; ++k) se += negbuf[k * NROWS + r];
    s += logf(se / (float)NROWS + EPS);
  }
  if (threadIdx.x < SEQLEN) s += TEMP * posPart[threadIdx.x] / (float)SEQLEN;
  __shared__ float sm[256];
  sm[threadIdx.x] = s;
  __syncthreads();
  for (int o = blockDim.x / 2; o > 0; o >>= 1) {
    if (threadIdx.x < o) sm[threadIdx.x] += sm[threadIdx.x + o];
    __syncthreads();
  }
  if (threadIdx.x == 0) out[0] = sm[0];
}

// ---------------------------------------------------------------------------
// Host launcher
// ---------------------------------------------------------------------------
extern "C" void kernel_launch(void* const* d_in, const int* in_sizes, int n_in,
                              void* d_out, int out_size, void* d_ws, size_t ws_size,
                              hipStream_t stream) {
  const int*   data    = (const int*)d_in[0];    // [128,64]
  const int*   samples = (const int*)d_in[1];    // [10,8192]
  const float* embW    = (const float*)d_in[2];  // [32000,1024]
  const float* W_ih    = (const float*)d_in[3];  // [1024,1024]
  const float* b_ih    = (const float*)d_in[4];  // [1024]
  const float* W_hh    = (const float*)d_in[5];  // [1024,1024]
  const float* b_hh    = (const float*)d_in[6];  // [1024]
  float* out = (float*)d_out;

  // Workspace carve-up (256-byte aligned)
  char* p = (char*)d_ws;
  auto take = [&](size_t bytes) {
    char* r = p;
    p += (bytes + 255) & ~(size_t)255;
    return (void*)r;
  };
  bf16_t* WihB  = (bf16_t*)take((size_t)NHID * NHID * sizeof(bf16_t));  // 2 MB
  bf16_t* WhhB  = (bf16_t*)take((size_t)NHID * NHID * sizeof(bf16_t));  // 2 MB
  float*  bsum  = (float*) take((size_t)NHID * sizeof(float));
  bf16_t* Embbf = (bf16_t*)take((size_t)NROWS * NHID * sizeof(bf16_t)); // 16 MB
  bf16_t* Hbf   = (bf16_t*)take((size_t)(SEQLEN + 1) * BSZ * NHID * sizeof(bf16_t)); // 17 MB
  float*  rawF  = (float*) take((size_t)(SEQLEN + 1) * BSZ * NHID * sizeof(float));  // 34 MB
  float*  hidU  = (float*) take((size_t)NROWS * NHID * sizeof(float));  // 34 MB
  float*  negbf = (float*) take((size_t)NS * NROWS * sizeof(float));    // 327 KB
  float*  posPt = (float*) take((size_t)SEQLEN * sizeof(float));

  // 1) weight/bias conversion to bf16 (layout preserved: W rows are B-frags)
  conv_w_kernel<<<(NHID * NHID + 255) / 256, 256, 0, stream>>>(
      W_ih, W_hh, b_ih, b_hh, WihB, WhhB, bsum);
  // 2) embedding gather + bf16 convert for the input sequence
  embed_kernel<<<NROWS, 256, 0, stream>>>(data, embW, Embbf);
  // 3) h0 = 0
  init_h0_kernel<<<(BSZ * NHID + 255) / 256, 256, 0, stream>>>(rawF, Hbf);
  // 4) sequential RNN scan (128 dependent launches; weights stay hot in L2)
  for (int t = 0; t < SEQLEN; ++t)
    rnn_step_kernel<<<32, 128, 0, stream>>>(Embbf, WihB, WhhB, bsum, Hbf, rawF, t);
  // 5) positive-term partials
  pos_kernel<<<SEQLEN, 256, 0, stream>>>(rawF, posPt);
  // 6) hiddens_U GEMM
  hidu_kernel<<<(NROWS / 64) * (NHID / 64), 128, 0, stream>>>(Hbf, WhhB, b_hh, hidU);
  // 7) negative term: gather-fused GEMM + tanh + distance + exp
  neg_kernel<<<NS * (NROWS / 16), 128, 0, stream>>>(
      samples, embW, WihB, b_ih, hidU, rawF, negbf);
  // 8) final scalar reduction
  final_kernel<<<1, 256, 0, stream>>>(posPt, negbf, out);
}
